// RelationshipAwareEncoder_18700287607434
// MI455X (gfx1250) — compile-verified, matrix operand order checked by reading, not run on hardware
//
#include <hip/hip_runtime.h>
#include <math.h>

#define N_SAMPLES 200000
#define D 128
#define EPS 1e-8f

typedef __attribute__((ext_vector_type(8)))  float        v8f;
typedef __attribute__((ext_vector_type(4)))  unsigned int v4u;
typedef __attribute__((ext_vector_type(8)))  unsigned int v8u;
typedef __attribute__((ext_vector_type(16))) __bf16       v16bf;

// ---------------- workspace layout (floats) ----------------
#define WS_CNT     0        // [128] sum obs_mask per column
#define WS_SUMMV   128      // [128] sum masked_values
#define WS_INTCNT  256      // [128] sum interventions
#define WS_SUMV    384      // [128] sum values
#define WS_SUMV2   512      // [128] sum values^2
#define WS_MEANS   640      // [128]
#define WS_SDIV    768      // [128] sample std
#define WS_DENOM   896      // [1]   sum(obs_mask[:,0]) + EPS
#define WS_X0      1024     // [128] layer0 pre-activation accumulator
#define WS_PHH     1152     // [16384] Chi^T Chi
#define WS_PHL     17536    // [16384] Chi^T Clo
#define WS_INTS    33920    // [16384] I^T (Vhi+Vlo)
#define WS_FEAT    50304    // [33024] feature vector
#define ZERO_COUNT 50304
#define F_TOTAL    33024

// ---------------- output layout (floats) ----------------
#define OUT_PROBS  0
#define OUT_SCORES 128
#define OUT_CORR   256
#define OUT_EFF    16640
#define OUT_TCORR  33024

// bf16 split helpers (RNE)
__device__ __forceinline__ unsigned short f2bf(float x) {
    unsigned int b = __builtin_bit_cast(unsigned int, x);
    b += 0x7FFFu + ((b >> 16) & 1u);
    return (unsigned short)(b >> 16);
}
__device__ __forceinline__ float bf2f(unsigned short h) {
    return __builtin_bit_cast(float, ((unsigned int)h) << 16);
}

// =========================================================
// K0: zero accumulators
// =========================================================
__global__ void k_zero(float* __restrict__ ws, int n) {
    int i = blockIdx.x * blockDim.x + threadIdx.x;
    if (i < n) ws[i] = 0.0f;
}

// =========================================================
// K1: column statistics, one streaming pass over data
// =========================================================
__global__ void k_stats(const float* __restrict__ data, float* __restrict__ ws) {
    int tid = threadIdx.x;                 // 256 threads = 2 rows x 128 cols
    int col = tid & (D - 1);
    int rofs = tid >> 7;
    float c_obs = 0.f, s_mv = 0.f, c_int = 0.f, s_v = 0.f, s_v2 = 0.f;
    for (int n = blockIdx.x * 2 + rofs; n < N_SAMPLES; n += gridDim.x * 2) {
        const float* p = data + (size_t)(n * D + col) * 3;
        float v = p[0], itv = p[1], obs = p[2];
        float om = obs * (1.0f - itv);
        float mv = (om > 0.0f) ? v : 0.0f;
        c_obs += om; s_mv += mv; c_int += itv; s_v += v; s_v2 += v * v;
    }
    atomicAdd(&ws[WS_CNT    + col], c_obs);
    atomicAdd(&ws[WS_SUMMV  + col], s_mv);
    atomicAdd(&ws[WS_INTCNT + col], c_int);
    atomicAdd(&ws[WS_SUMV   + col], s_v);
    atomicAdd(&ws[WS_SUMV2  + col], s_v2);
}

// =========================================================
// K2: finalize means / std / denominator  (1 block, 128 threads)
// =========================================================
__global__ void k_finalize_stats(float* __restrict__ ws) {
    int d = threadIdx.x;
    float counts = ws[WS_CNT + d] + EPS;
    ws[WS_MEANS + d] = ws[WS_SUMMV + d] / counts;
    float ev  = ws[WS_SUMV  + d] / (float)N_SAMPLES;
    float ev2 = ws[WS_SUMV2 + d] / (float)N_SAMPLES;
    float var = ev2 - ev * ev;
    ws[WS_SDIV + d] = sqrtf(var > 0.0f ? var : 0.0f);
    if (d == 0) ws[WS_DENOM] = ws[WS_CNT + 0] + EPS;
}

// =========================================================
// K3: split-bf16 streaming GEMM via V_WMMA_F32_16X16X32_BF16
//   Phh  += Chi^T Chi        Phl += Chi^T Clo     (cov = Phh+Phl+Phl^T)
//   intS += I^T Vhi + I^T Vlo
// 512 threads = 16 waves; wave owns 16 rows x 64 cols of output.
// LDS: bf16 tiles stored column-major [col][k], stride 40 (16B aligned,
// bank-conflict-free b128 fragment loads per ISA 16-bit A/B layout).
// =========================================================
#define STR 40                       // bf16 elements per column row

__device__ __forceinline__ v16bf load_frag(const unsigned short* ls, int colidx, int kb) {
    const unsigned short* p = ls + colidx * STR + kb;
    v4u lo = *(const v4u*)(p);        // K = kb .. kb+7   (pair-packed)
    v4u hi = *(const v4u*)(p + 16);   // K = kb+16 .. kb+23
    v8u j = {lo.x, lo.y, lo.z, lo.w, hi.x, hi.y, hi.z, hi.w};
    return __builtin_bit_cast(v16bf, j);
}

__global__ void __launch_bounds__(512)
k_gemm(const float* __restrict__ data, float* __restrict__ ws) {
    __shared__ __align__(16) unsigned short ls_chi[D * STR];
    __shared__ __align__(16) unsigned short ls_clo[D * STR];
    __shared__ __align__(16) unsigned short ls_int[D * STR];
    __shared__ __align__(16) unsigned short ls_vhi[D * STR];
    __shared__ __align__(16) unsigned short ls_vlo[D * STR];

    int tid  = threadIdx.x;
    int lane = tid & 31;
    int wave = tid >> 5;                  // 0..15
    int col  = tid & (D - 1);
    int rgrp = tid >> 7;                  // 0..3
    float mean_c = ws[WS_MEANS + col];

    v8f ahh[4], ahl[4], aiv[4];
#pragma unroll
    for (int t = 0; t < 4; t++) {
        ahh[t] = (v8f){0.f,0.f,0.f,0.f,0.f,0.f,0.f,0.f};
        ahl[t] = (v8f){0.f,0.f,0.f,0.f,0.f,0.f,0.f,0.f};
        aiv[t] = (v8f){0.f,0.f,0.f,0.f,0.f,0.f,0.f,0.f};
    }

    const int nChunks = (N_SAMPLES + 31) / 32;    // 6250
    const int mb    = (wave & 7) * 16;            // M strip
    const int nbase = (wave >> 3) * 64;           // N half
    const int li    = lane & 15;
    const int kb    = (lane < 16) ? 0 : 8;        // ISA 16-bit frag K base

    for (int ch = blockIdx.x; ch < nChunks; ch += gridDim.x) {
        int base = ch * 32;
        // ---- stage 32 rows: split to bf16 hi/lo, pack 2 rows per dword ----
#pragma unroll
        for (int j = 0; j < 4; j++) {
            int r  = rgrp * 2 + j * 8;            // even row of the pair
            int n0 = base + r;
            const float* p = data + (size_t)(n0 * D + col) * 3;
            float v0 = 0.f, i0 = 0.f, o0 = 0.f, v1 = 0.f, i1 = 0.f, o1 = 0.f;
            if (n0 < N_SAMPLES)     { v0 = p[0];       i0 = p[1];       o0 = p[2];       }
            if (n0 + 1 < N_SAMPLES) { v1 = p[3 * D];   i1 = p[3 * D + 1]; o1 = p[3 * D + 2]; }
            float om0 = o0 * (1.f - i0), om1 = o1 * (1.f - i1);
            float c0 = (((om0 > 0.f) ? v0 : 0.f) - mean_c) * om0;
            float c1 = (((om1 > 0.f) ? v1 : 0.f) - mean_c) * om1;
            unsigned short ch0 = f2bf(c0), ch1 = f2bf(c1);
            unsigned short cl0 = f2bf(c0 - bf2f(ch0)), cl1 = f2bf(c1 - bf2f(ch1));
            unsigned short vh0 = f2bf(v0), vh1 = f2bf(v1);
            unsigned short vl0 = f2bf(v0 - bf2f(vh0)), vl1 = f2bf(v1 - bf2f(vh1));
            unsigned short it0 = f2bf(i0), it1 = f2bf(i1);
            unsigned int di = (unsigned)(col * STR + r) >> 1;
            ((unsigned int*)ls_chi)[di] = (unsigned)ch0 | ((unsigned)ch1 << 16);
            ((unsigned int*)ls_clo)[di] = (unsigned)cl0 | ((unsigned)cl1 << 16);
            ((unsigned int*)ls_int)[di] = (unsigned)it0 | ((unsigned)it1 << 16);
            ((unsigned int*)ls_vhi)[di] = (unsigned)vh0 | ((unsigned)vh1 << 16);
            ((unsigned int*)ls_vlo)[di] = (unsigned)vl0 | ((unsigned)vl1 << 16);
        }
        // prefetch next chunk this block will touch
        {
            int nn = base + gridDim.x * 32 + rgrp * 2;
            if (nn < N_SAMPLES)
                __builtin_prefetch(data + (size_t)(nn * D + col) * 3, 0, 1);
        }
        __syncthreads();

        // ---- one K=32 step: 4 bf16 WMMA per N-tile ----
        v16bf a_c = load_frag(ls_chi, mb + li, kb);
        v16bf a_i = load_frag(ls_int, mb + li, kb);
#pragma unroll
        for (int nt = 0; nt < 4; nt++) {
            int c = nbase + nt * 16 + li;
            v16bf b_ch = load_frag(ls_chi, c, kb);
            ahh[nt] = __builtin_amdgcn_wmma_f32_16x16x32_bf16(
                false, a_c, false, b_ch, (short)0, ahh[nt], false, false);
            v16bf b_cl = load_frag(ls_clo, c, kb);
            ahl[nt] = __builtin_amdgcn_wmma_f32_16x16x32_bf16(
                false, a_c, false, b_cl, (short)0, ahl[nt], false, false);
            v16bf b_vh = load_frag(ls_vhi, c, kb);
            aiv[nt] = __builtin_amdgcn_wmma_f32_16x16x32_bf16(
                false, a_i, false, b_vh, (short)0, aiv[nt], false, false);
            v16bf b_vl = load_frag(ls_vlo, c, kb);
            aiv[nt] = __builtin_amdgcn_wmma_f32_16x16x32_bf16(
                false, a_i, false, b_vl, (short)0, aiv[nt], false, false);
        }
        __syncthreads();
    }

    // ---- flush: C/D layout VGPR r -> M=r (lanes 0-15) / M=r+8 (16-31) ----
    float* Phh  = ws + WS_PHH;
    float* Phl  = ws + WS_PHL;
    float* intS = ws + WS_INTS;
    int mrow = mb + ((lane < 16) ? 0 : 8);
#pragma unroll
    for (int nt = 0; nt < 4; nt++) {
        int ncol = nbase + nt * 16 + li;
#pragma unroll
        for (int r = 0; r < 8; r++) {
            atomicAdd(&Phh [(mrow + r) * D + ncol], ahh[nt][r]);
            atomicAdd(&Phl [(mrow + r) * D + ncol], ahl[nt][r]);
            atomicAdd(&intS[(mrow + r) * D + ncol], aiv[nt][r]);
        }
    }
}

// =========================================================
// K4: corr / effects + feature vector      (64 blocks x 256)
//   cov = (Phh + Phl + Phl^T) / denom
// =========================================================
__global__ void k_finalize_mats(float* __restrict__ ws, float* __restrict__ out) {
    int idx = blockIdx.x * blockDim.x + threadIdx.x;   // 0..16383
    float* Phh  = ws + WS_PHH;
    float* Phl  = ws + WS_PHL;
    float* intS = ws + WS_INTS;
    float* feat = ws + WS_FEAT;
    float denomc = ws[WS_DENOM];
    int i = idx >> 7, j = idx & (D - 1);

    float cov = (Phh[idx] + Phl[idx] + Phl[j * D + i]) / denomc;
    float di  = (Phh[i * (D + 1)] + 2.0f * Phl[i * (D + 1)]) / denomc;
    float dj  = (Phh[j * (D + 1)] + 2.0f * Phl[j * (D + 1)]) / denomc;
    float corr = cov / (sqrtf(di + EPS) * sqrtf(dj + EPS));
    out[OUT_CORR + idx] = corr;
    feat[idx] = corr;

    float ic = ws[WS_INTCNT + i];
    float eff = 0.0f;
    if (ic > 0.0f && i != j)
        eff = intS[idx] / fmaxf(ic, 1.0f) - ws[WS_MEANS + j];
    out[OUT_EFF + idx] = eff;
    feat[16384 + idx] = eff;

    if (idx < D) {
        feat[32768 + idx] = ws[WS_SDIV + idx];
        feat[32896 + idx] = ws[WS_MEANS + idx];
    }
}

// =========================================================
// K5: x0raw = all_features @ W0   (129 blocks x 128 threads)
// =========================================================
__global__ void k_mlp0(const float* __restrict__ W0, float* __restrict__ ws) {
    const float* feat = ws + WS_FEAT;
    int h = threadIdx.x;
    int f0 = blockIdx.x * 256;
    int f1 = f0 + 256; if (f1 > F_TOTAL) f1 = F_TOTAL;
    float acc = 0.0f;
    for (int f = f0; f < f1; f++)
        acc += feat[f] * W0[(size_t)f * D + h];
    atomicAdd(&ws[WS_X0 + h], acc);
}

// =========================================================
// K6: MLP head + parent scores + softmax   (1 block, 128 threads)
// =========================================================
__device__ __forceinline__ float blk_sum(float v, float* red) {
    int t = threadIdx.x;
    red[t] = v; __syncthreads();
    for (int s = 64; s > 0; s >>= 1) { if (t < s) red[t] += red[t + s]; __syncthreads(); }
    float r = red[0]; __syncthreads(); return r;
}
__device__ __forceinline__ float blk_max(float v, float* red) {
    int t = threadIdx.x;
    red[t] = v; __syncthreads();
    for (int s = 64; s > 0; s >>= 1) { if (t < s) red[t] = fmaxf(red[t], red[t + s]); __syncthreads(); }
    float r = red[0]; __syncthreads(); return r;
}

__global__ void k_head(const float* __restrict__ b0,
                       const float* __restrict__ W1, const float* __restrict__ b1,
                       const float* __restrict__ g1, const float* __restrict__ o1,
                       const float* __restrict__ W2, const float* __restrict__ b2,
                       const float* __restrict__ g2, const float* __restrict__ o2,
                       const float* __restrict__ Wp, const float* __restrict__ bp,
                       const int* __restrict__ tptr,
                       float* __restrict__ ws, float* __restrict__ out) {
    __shared__ float sx[D], sy[D], red[D], tf[3 * D];
    int t = threadIdx.x;
    int tgt = *tptr;

    sx[t] = fmaxf(ws[WS_X0 + t] + b0[t], 0.0f);
    __syncthreads();

    float acc = b1[t];
    for (int k = 0; k < D; k++) acc += sx[k] * W1[k * D + t];
    float mu  = blk_sum(acc, red) * (1.0f / D);
    float dd  = acc - mu;
    float var = blk_sum(dd * dd, red) * (1.0f / D);
    float xn  = dd * rsqrtf(var + 1e-5f) * g1[t] + o1[t];
    sy[t] = fmaxf(xn, 0.0f);
    __syncthreads();

    acc = b2[t];
    for (int k = 0; k < D; k++) acc += sy[k] * W2[k * D + t];
    mu  = blk_sum(acc, red) * (1.0f / D);
    dd  = acc - mu;
    var = blk_sum(dd * dd, red) * (1.0f / D);
    xn  = dd * rsqrtf(var + 1e-5f) * g2[t] + o2[t];
    sx[t] = fmaxf(xn, 0.0f);
    __syncthreads();

    float tc = out[OUT_CORR + tgt * D + t];
    tf[t]         = tc;
    tf[D + t]     = out[OUT_EFF + t * D + tgt];
    tf[2 * D + t] = sx[t];
    out[OUT_TCORR + t] = tc;
    __syncthreads();

    float sc = bp[t];
    for (int f = 0; f < 3 * D; f++) sc += tf[f] * Wp[f * D + t];
    out[OUT_SCORES + t] = sc;

    float m  = (t == tgt) ? -1e9f : sc;
    float mx = blk_max(m, red);
    float e  = __expf(m - mx);
    float s  = blk_sum(e, red);
    out[OUT_PROBS + t] = e / s;
}

// =========================================================
extern "C" void kernel_launch(void* const* d_in, const int* in_sizes, int n_in,
                              void* d_out, int out_size, void* d_ws, size_t ws_size,
                              hipStream_t stream) {
    const float* data = (const float*)d_in[0];
    const float* W0   = (const float*)d_in[1];
    const float* b0   = (const float*)d_in[2];
    const float* W1   = (const float*)d_in[3];
    const float* b1   = (const float*)d_in[4];
    const float* g1   = (const float*)d_in[5];
    const float* o1   = (const float*)d_in[6];
    const float* W2   = (const float*)d_in[7];
    const float* b2   = (const float*)d_in[8];
    const float* g2   = (const float*)d_in[9];
    const float* o2   = (const float*)d_in[10];
    const float* Wp   = (const float*)d_in[11];
    const float* bp   = (const float*)d_in[12];
    const int*   tgt  = (const int*)d_in[13];

    float* ws  = (float*)d_ws;
    float* out = (float*)d_out;

    k_zero<<<(ZERO_COUNT + 255) / 256, 256, 0, stream>>>(ws, ZERO_COUNT);
    k_stats<<<512, 256, 0, stream>>>(data, ws);
    k_finalize_stats<<<1, 128, 0, stream>>>(ws);
    k_gemm<<<208, 512, 0, stream>>>(data, ws);
    k_finalize_mats<<<64, 256, 0, stream>>>(ws, out);
    k_mlp0<<<(F_TOTAL + 255) / 256, 128, 0, stream>>>(W0, ws);
    k_head<<<1, 128, 0, stream>>>(b0, W1, b1, g1, o1, W2, b2, g2, o2, Wp, bp,
                                  tgt, ws, out);
}
